// EquivariantGNN_60919816127138
// MI455X (gfx1250) — matrix operand order repaired and checked by reading, not compiled, not run on hardware
//
#include <hip/hip_runtime.h>
#include <hip/hip_bf16.h>
#include <math.h>

typedef __attribute__((ext_vector_type(16))) _Float16 v16h;
typedef __attribute__((ext_vector_type(8)))  _Float16 v8h;
typedef __attribute__((ext_vector_type(4)))  _Float16 v4h;
typedef __attribute__((ext_vector_type(8)))  float    v8f;

#define N_NODES  50000
#define N_EDGES  400000
#define CCH      32
#define HID      64
#define N_GRAPHS 64
#define R_MAXF   5.0f
#define EPSF     1e-5f

__device__ __forceinline__ float sigm(float x){
  // fast sigmoid: v_exp_f32 + v_rcp_f32 (no IEEE divide chain)
  return __builtin_amdgcn_rcpf(1.0f + __expf(-x));
}
__device__ __forceinline__ float silu_f(float x){ return x*sigm(x); }

static __device__ __forceinline__ v8f wmma_f16(v16h a, v16h b, v8f c){
  // D = A(16x32 f16) x B(32x16 f16) + C(16x16 f32)
  return __builtin_amdgcn_wmma_f32_16x16x32_f16(false, a, false, b, (short)0, c, false, false);
}

// A fragment reload from LDS stage row (row-major [16][72], 16B-aligned chunks)
static __device__ __forceinline__ v16h load_A_frag(const _Float16* row, int base, int hi){
  const int kb = base + 16*hi;
  v8h lo = *(const v8h*)(row + kb);
  v8h hp = *(const v8h*)(row + kb + 8);
  return __builtin_shufflevector(lo, hp, 0,1,2,3,4,5,6,7,8,9,10,11,12,13,14,15);
}

// B fragment from n-major transposed f16 weights (row stride 64 halfs)
static __device__ __forceinline__ v16h load_B_frag(const _Float16* wt, int n, int kb){
  return *(const v16h*)(wt + n*64 + kb);
}

// ---------------------------------------------------------------------------
// Per-wave LDS scratch for the fused edge kernel
// ---------------------------------------------------------------------------
struct __align__(16) WaveLds {
  _Float16 stage[16][72];   // h1/h2 staging; stride 144B -> 16B-aligned subchunks
  _Float16 wrow [16][132];  // TP weights, interleaved: [edge][chan*4 + path]
  float    meta [16][8];    // y0,y1,y2, src(bits), dst(bits); 32B rows
};

// ---------------------------------------------------------------------------
// Fused edge kernel: geometry -> Bessel basis -> 3-stage radial MLP (WMMA)
//   -> tensor-product messages -> atomic scatter to node aggregates.
// One wave32 = 16 edges.  8 waves / block.
// ---------------------------------------------------------------------------
__launch_bounds__(256)
__global__ void edge_layer_kernel(
    const float* __restrict__ pos,
    const int*   __restrict__ eidx,     // (2,E): row0=src, row1=dst
    const float* __restrict__ freqs,    // (8,)
    const _Float16* __restrict__ w1t, const float* __restrict__ rb1,  // [64][8]  n-major
    const _Float16* __restrict__ w2t, const float* __restrict__ rb2,  // [64][64] n-major
    const _Float16* __restrict__ w3t, const float* __restrict__ rb3,  // [128][64] n-major
    const float* __restrict__ sfeat,    // (N,32)
    const float* __restrict__ vfeat,    // (N,32,3)
    float* __restrict__ agg_sf,
    float* __restrict__ agg_sg,
    float* __restrict__ agg_v)
{
  __shared__ WaveLds smem[8];
  const int lane = threadIdx.x & 31;
  const int wave = threadIdx.x >> 5;
  WaveLds& W = smem[wave];
  const int m  = lane & 15;     // N column (GEMM phases) / edge row (geometry)
  const int hi = lane >> 4;     // lane half
  const int e  = (blockIdx.x * 8 + wave) * 16 + m;

  // ---- geometry (both lane halves compute the same edge; cheap) ----
  const int si = eidx[e];
  const int di = eidx[N_EDGES + e];
  const float dx = pos[di*3+0] - pos[si*3+0];
  const float dy = pos[di*3+1] - pos[si*3+1];
  const float dz = pos[di*3+2] - pos[si*3+2];
  const float dist = sqrtf(dx*dx + dy*dy + dz*dz);
  const float invY = __builtin_amdgcn_rcpf(fmaxf(dist, 1e-9f));
  float x  = fminf(dist * (1.0f / R_MAXF), 1.0f);
  float x2 = x*x, x3 = x2*x;
  float cut = 1.0f - 6.0f*x3*x2 + 15.0f*x2*x2 - 10.0f*x3;
  if (dist > R_MAXF) cut = 0.0f;
  const float rinv = __builtin_amdgcn_rcpf(fmaxf(dist, 1e-6f));

  if (hi == 0) {
    float4 q;
    q.x = dx*invY; q.y = dy*invY; q.z = dz*invY;
    q.w = __int_as_float(si);
    *(float4*)(&W.meta[m][0]) = q;
    W.meta[m][4] = __int_as_float(di);
  }

  // ---- A1 fragment: basis (16 edges x 32K, only K=0..7 nonzero) ----
  v16h a1;
  #pragma unroll
  for (int i = 0; i < 16; ++i) a1[i] = (_Float16)0.0f;
  if (hi == 0) {
    #pragma unroll
    for (int k = 0; k < 8; ++k)
      a1[k] = (_Float16)(__sinf(freqs[k] * dist) * rinv * cut);
  }

  v8h z8;
  #pragma unroll
  for (int i = 0; i < 8; ++i) z8[i] = (_Float16)0.0f;

  // ---- GEMM1: h1 = silu(basis @ W1 + b1)  -> stage (D-layout store) ----
  #pragma unroll
  for (int t = 0; t < 4; ++t) {
    const int n = t*16 + m;
    v8h lo = z8;
    if (hi == 0) lo = *(const v8h*)(w1t + n*8);   // 8 contiguous halfs, 16B
    v16h b = __builtin_shufflevector(lo, z8, 0,1,2,3,4,5,6,7,8,9,10,11,12,13,14,15);
    v8f acc = {0.f,0.f,0.f,0.f,0.f,0.f,0.f,0.f};
    acc = wmma_f16(a1, b, acc);
    const float bias = rb1[n];
    #pragma unroll
    for (int r = 0; r < 8; ++r)
      W.stage[r + 8*hi][n] = (_Float16)silu_f(acc[r] + bias);
  }
  __syncthreads();

  // ---- re-layout h1 (LDS) -> A2 fragments, two k-steps of 32 ----
  const _Float16* srow = &W.stage[m][0];
  v16h a20 = load_A_frag(srow, 0, hi);
  v16h a21 = load_A_frag(srow, 32, hi);
  __syncthreads();

  // ---- GEMM2: h2 = silu(h1 @ W2 + b2) -> stage (overwrite) ----
  #pragma unroll
  for (int t = 0; t < 4; ++t) {
    const int n = t*16 + m;
    v8f acc = {0.f,0.f,0.f,0.f,0.f,0.f,0.f,0.f};
    acc = wmma_f16(a20, load_B_frag(w2t, n, 16*hi),      acc);
    acc = wmma_f16(a21, load_B_frag(w2t, n, 32 + 16*hi), acc);
    const float bias = rb2[n];
    #pragma unroll
    for (int r = 0; r < 8; ++r)
      W.stage[r + 8*hi][n] = (_Float16)silu_f(acc[r] + bias);
  }
  __syncthreads();

  // ---- re-layout h2 -> A3 fragments ----
  v16h a30 = load_A_frag(srow, 0, hi);
  v16h a31 = load_A_frag(srow, 32, hi);

  // ---- GEMM3: w = h2 @ W3 + b3 -> wrow (interleaved [edge][chan*4+path]) ----
  #pragma unroll
  for (int t = 0; t < 8; ++t) {
    const int n  = t*16 + m;
    const int ci = (t & 1)*16 + m;   // channel = n & 31
    const int p  = t >> 1;           // TP path  = n >> 5
    v8f acc = {0.f,0.f,0.f,0.f,0.f,0.f,0.f,0.f};
    acc = wmma_f16(a30, load_B_frag(w3t, n, 16*hi),      acc);
    acc = wmma_f16(a31, load_B_frag(w3t, n, 32 + 16*hi), acc);
    const float bias = rb3[n];
    #pragma unroll
    for (int r = 0; r < 8; ++r)
      W.wrow[r + 8*hi][ci*4 + p] = (_Float16)(acc[r] + bias);
  }
  __syncthreads();

  // ---- tensor-product messages + coalesced atomic scatter ----
  // lanes = 32 channels; loop over the wave's 16 edges
  const int c = lane;
  for (int ei = 0; ei < 16; ++ei) {
    const float4 q = *(const float4*)(&W.meta[ei][0]);   // ds_load_b128
    const int   sn = __float_as_int(q.w);
    const int   dn = __float_as_int(W.meta[ei][4]);      // ds_load_b32
    const v4h  wv  = *(const v4h*)(&W.wrow[ei][c*4]);    // ds_load_b64
    const float w1 = (float)wv[0];
    const float w2 = (float)wv[1];
    const float w3 = (float)wv[2];
    const float w4 = (float)wv[3];
    const float xs = sfeat[sn*CCH + c];
    const float* xvp = vfeat + (size_t)(sn*CCH + c)*3;
    const float xv0 = xvp[0], xv1 = xvp[1], xv2 = xvp[2];

    atomicAdd(agg_sf + dn*CCH + c, w1 * xs);
    atomicAdd(agg_sg + dn*CCH + c, w2 * (xv0*q.x + xv1*q.y + xv2*q.z));
    float* avp = agg_v + (size_t)(dn*CCH + c)*3;
    atomicAdd(avp + 0, w3*xs*q.x + w4*xv0);
    atomicAdd(avp + 1, w3*xs*q.y + w4*xv1);
    atomicAdd(avp + 2, w3*xs*q.z + w4*xv2);
  }
}

// ---------------------------------------------------------------------------
// Node pass 1: gated nonlinearity + self-interaction + BN statistics.
// One wave32 per node (lane = channel).  8 nodes / block.
// ---------------------------------------------------------------------------
__launch_bounds__(256)
__global__ void node_kernel1(
    const float* __restrict__ agg_sf, const float* __restrict__ agg_sg,
    const float* __restrict__ agg_v,
    const float* __restrict__ Wss, const float* __restrict__ bss,
    const float* __restrict__ Wvv,
    float* __restrict__ s_pre, float* __restrict__ v_pre,
    float* __restrict__ stats)                   // [96]: sum_s, sum_s2, sum_v2
{
  __shared__ float red[96];
  if (threadIdx.x < 96) red[threadIdx.x] = 0.0f;
  __syncthreads();

  const int wave = threadIdx.x >> 5;
  const int c    = threadIdx.x & 31;
  const int node = blockIdx.x * 8 + wave;
  if (node < N_NODES) {
    const float sf = agg_sf[node*CCH + c];
    const float sg = agg_sg[node*CCH + c];
    const float* ap = agg_v + (size_t)(node*CCH + c)*3;
    const float ga = sigm(sg);
    const float sa = silu_f(sf);
    const float va0 = ga*ap[0], va1 = ga*ap[1], va2 = ga*ap[2];

    float sn  = bss[c];
    float vn0 = 0.f, vn1 = 0.f, vn2 = 0.f;
    #pragma unroll
    for (int k = 0; k < 32; ++k) {
      const float sk = __shfl(sa, k, 32);
      const float ws = Wss[k*CCH + c];
      const float wv = Wvv[k*CCH + c];
      sn  += sk * ws;
      vn0 += __shfl(va0, k, 32) * wv;
      vn1 += __shfl(va1, k, 32) * wv;
      vn2 += __shfl(va2, k, 32) * wv;
    }
    s_pre[node*CCH + c] = sn;
    float* vp = v_pre + (size_t)(node*CCH + c)*3;
    vp[0] = vn0; vp[1] = vn1; vp[2] = vn2;

    atomicAdd(&red[c],      sn);
    atomicAdd(&red[32 + c], sn*sn);
    atomicAdd(&red[64 + c], vn0*vn0 + vn1*vn1 + vn2*vn2);
  }
  __syncthreads();
  if (threadIdx.x < 96) atomicAdd(&stats[threadIdx.x], red[threadIdx.x]);
}

// ---------------------------------------------------------------------------
// Node pass 2: finalize e3nn batch-norm + residual add into s, v.
// ---------------------------------------------------------------------------
__launch_bounds__(256)
__global__ void node_kernel2(
    const float* __restrict__ s_pre, const float* __restrict__ v_pre,
    const float* __restrict__ stats,
    const float* __restrict__ gs, const float* __restrict__ gb,
    const float* __restrict__ gv,
    float* __restrict__ s, float* __restrict__ v)
{
  const int idx = blockIdx.x * 256 + threadIdx.x;  // over N*32
  if (idx >= N_NODES*CCH) return;
  const int c = idx & 31;
  const float inv_n = 1.0f / (float)N_NODES;
  const float mu   = stats[c] * inv_n;
  const float var  = stats[32 + c] * inv_n - mu*mu;
  const float sstd = rsqrtf(var + EPSF);
  const float vstd = rsqrtf(stats[64 + c] * inv_n * (1.0f/3.0f) + EPSF);

  s[idx] += (s_pre[idx] - mu) * sstd * gs[c] + gb[c];
  const float scale = vstd * gv[c];
  v[(size_t)idx*3 + 0] += v_pre[(size_t)idx*3 + 0] * scale;
  v[(size_t)idx*3 + 1] += v_pre[(size_t)idx*3 + 1] * scale;
  v[(size_t)idx*3 + 2] += v_pre[(size_t)idx*3 + 2] * scale;
}

// ---------------------------------------------------------------------------
__launch_bounds__(256)
__global__ void init_s_kernel(const int* __restrict__ species,
                              const float* __restrict__ emb,
                              float* __restrict__ s)
{
  const int idx = blockIdx.x * 256 + threadIdx.x;
  if (idx < N_NODES*CCH) {
    const int n = idx >> 5, c = idx & 31;
    s[idx] = emb[species[n]*CCH + c];
  }
}

// f32 [K][N] -> f16 [N][K] (n-major so B fragments are contiguous)
__launch_bounds__(256)
__global__ void transpose_f16_kernel(const float* __restrict__ src,
                                     _Float16* __restrict__ dst, int K, int N)
{
  const int i = blockIdx.x * 256 + threadIdx.x;
  if (i < K*N) {
    const int k = i / N, n = i - k*N;
    dst[n*K + k] = (_Float16)src[i];
  }
}

// ---------------------------------------------------------------------------
// Readout: per-node energy -> per-graph energy (LDS pre-aggregation; batch is
// sorted so each block touches few graphs).  One wave32 per node.
// ---------------------------------------------------------------------------
__launch_bounds__(256)
__global__ void out_kernel(const float* __restrict__ s,
                           const int* __restrict__ batch,
                           const int* __restrict__ species,
                           const float* __restrict__ W_out,
                           const float* __restrict__ b_out,
                           const float* __restrict__ atom_ref,
                           float* __restrict__ out)
{
  __shared__ float g[N_GRAPHS];
  if (threadIdx.x < N_GRAPHS) g[threadIdx.x] = 0.0f;
  __syncthreads();

  const int wave = threadIdx.x >> 5;
  const int c    = threadIdx.x & 31;
  const int node = blockIdx.x * 8 + wave;
  if (node < N_NODES) {
    float val = s[node*CCH + c] * W_out[c];
    #pragma unroll
    for (int off = 16; off > 0; off >>= 1) val += __shfl_xor(val, off, 32);
    if (c == 0)
      atomicAdd(&g[batch[node]], val + b_out[0] + atom_ref[species[node]]);
  }
  __syncthreads();
  if (threadIdx.x < N_GRAPHS) {
    const float x = g[threadIdx.x];
    if (x != 0.0f) atomicAdd(out + threadIdx.x, x);
  }
}

// ---------------------------------------------------------------------------
extern "C" void kernel_launch(void* const* d_in, const int* in_sizes, int n_in,
                              void* d_out, int out_size, void* d_ws, size_t ws_size,
                              hipStream_t stream)
{
  const int*   species  = (const int*)  d_in[0];
  const float* pos      = (const float*)d_in[1];
  const int*   eidx     = (const int*)  d_in[2];
  const int*   batch    = (const int*)  d_in[3];
  const float* emb      = (const float*)d_in[4];
  const float* freqs    = (const float*)d_in[5];
  const float* rW1      = (const float*)d_in[6];
  const float* rb1      = (const float*)d_in[7];
  const float* rW2      = (const float*)d_in[8];
  const float* rb2      = (const float*)d_in[9];
  const float* rW3      = (const float*)d_in[10];
  const float* rb3      = (const float*)d_in[11];
  const float* Wss      = (const float*)d_in[12];
  const float* bss      = (const float*)d_in[13];
  const float* Wvv      = (const float*)d_in[14];
  const float* bn_gs    = (const float*)d_in[15];
  const float* bn_bs    = (const float*)d_in[16];
  const float* bn_gv    = (const float*)d_in[17];
  const float* W_out    = (const float*)d_in[18];
  const float* b_out    = (const float*)d_in[19];
  const float* atom_ref = (const float*)d_in[20];
  (void)in_sizes; (void)n_in; (void)out_size; (void)ws_size;

  char* ws = (char*)d_ws;
  size_t off = 0;
  auto alloc = [&](size_t bytes) -> void* {
    void* p = ws + off;
    off += (bytes + 255) & ~(size_t)255;
    return p;
  };
  float*    s      = (float*)   alloc((size_t)N_NODES*CCH*4);
  float*    v      = (float*)   alloc((size_t)N_NODES*CCH*3*4);
  float*    agg_sf = (float*)   alloc((size_t)N_NODES*CCH*4);
  float*    agg_sg = (float*)   alloc((size_t)N_NODES*CCH*4);
  float*    agg_v  = (float*)   alloc((size_t)N_NODES*CCH*3*4);
  float*    s_pre  = (float*)   alloc((size_t)N_NODES*CCH*4);
  float*    v_pre  = (float*)   alloc((size_t)N_NODES*CCH*3*4);
  _Float16* w1t    = (_Float16*)alloc((size_t)2*8*64*2);
  _Float16* w2t    = (_Float16*)alloc((size_t)2*64*64*2);
  _Float16* w3t    = (_Float16*)alloc((size_t)2*64*128*2);
  float*    stats  = (float*)   alloc(96*4);

  // Weight f32 -> f16 transposed (n-major) conversion
  for (int l = 0; l < 2; ++l) {
    transpose_f16_kernel<<<(8*64   + 255)/256, 256, 0, stream>>>(rW1 + l*8*64,   w1t + l*8*64,   8,  64);
    transpose_f16_kernel<<<(64*64  + 255)/256, 256, 0, stream>>>(rW2 + l*64*64,  w2t + l*64*64,  64, 64);
    transpose_f16_kernel<<<(64*128 + 255)/256, 256, 0, stream>>>(rW3 + l*64*128, w3t + l*64*128, 64, 128);
  }

  init_s_kernel<<<(N_NODES*CCH + 255)/256, 256, 0, stream>>>(species, emb, s);
  hipMemsetAsync(v, 0, (size_t)N_NODES*CCH*3*4, stream);

  for (int l = 0; l < 2; ++l) {
    hipMemsetAsync(agg_sf, 0, (size_t)N_NODES*CCH*4,   stream);
    hipMemsetAsync(agg_sg, 0, (size_t)N_NODES*CCH*4,   stream);
    hipMemsetAsync(agg_v,  0, (size_t)N_NODES*CCH*3*4, stream);
    hipMemsetAsync(stats,  0, 96*4,                    stream);

    edge_layer_kernel<<<N_EDGES/128, 256, 0, stream>>>(
        pos, eidx, freqs,
        w1t + (size_t)l*8*64,   rb1 + l*64,
        w2t + (size_t)l*64*64,  rb2 + l*64,
        w3t + (size_t)l*64*128, rb3 + l*128,
        s, v, agg_sf, agg_sg, agg_v);

    node_kernel1<<<(N_NODES + 7)/8, 256, 0, stream>>>(
        agg_sf, agg_sg, agg_v,
        Wss + l*CCH*CCH, bss + l*CCH, Wvv + l*CCH*CCH,
        s_pre, v_pre, stats);

    node_kernel2<<<(N_NODES*CCH + 255)/256, 256, 0, stream>>>(
        s_pre, v_pre, stats,
        bn_gs + l*CCH, bn_bs + l*CCH, bn_gv + l*CCH,
        s, v);
  }

  hipMemsetAsync(d_out, 0, N_GRAPHS*4, stream);
  out_kernel<<<(N_NODES + 7)/8, 256, 0, stream>>>(
      s, batch, species, W_out, b_out, atom_ref, (float*)d_out);
}
